// MultiHeadAttention_10385230922526
// MI455X (gfx1250) — matrix operand order repaired and checked by reading, not compiled, not run on hardware
//
#include <hip/hip_runtime.h>

typedef __bf16 bf16_t;
typedef __attribute__((ext_vector_type(16))) __bf16 v16bf;
typedef __attribute__((ext_vector_type(8)))  float  v8f;

__device__ __forceinline__ v8f wmma_bf16(v16bf a, v16bf b, v8f c) {
  // 8 args: (neg_a, A, neg_b, B, c_mod, C, reuse_a, reuse_b)
  return __builtin_amdgcn_wmma_f32_16x16x32_bf16(false, a, false, b, (short)0, c,
                                                 false, false);
}

// A-fragment (16-bit, 16x32) element -> K index for lane-group grp (lane>>4).
// Contiguous order: element e holds K = grp*16 + e (matches the B fills).
__device__ __forceinline__ int a_kidx(int e, int grp) {
  return grp * 16 + e;
}

#define BATCH 4
#define SEQ   1024
#define EMB   1024
#define NH    16
#define HD    64
#define MTOT  (BATCH * SEQ)

// ------------------------------------------------------------------
// GEMM: C[M,N] = A[M,K](f32) @ W[N,K]^T(f32) + bias[N]
//   reorder=1: write bf16 into head-major [B,H,S,D] workspace
//   reorder=0: write fp32 row-major [M,N]
// Block tile 128x64, 8 waves, each wave a 32x32 register tile = 2x2 WMMA.
// ------------------------------------------------------------------
#define BM 128
#define BN 64
#define BK 32

__global__ __launch_bounds__(256)
void gemm_bias_wmma(const float* __restrict__ A,
                    const float* __restrict__ W,
                    const float* __restrict__ bias,
                    float* __restrict__ outF,
                    bf16_t* __restrict__ outHeads,
                    int reorder)
{
  __shared__ bf16_t As[BM][BK];   // 8 KB
  __shared__ bf16_t Ws[BN][BK];   // 4 KB

  const int tid  = threadIdx.x;
  const int lane = tid & 31, wave = tid >> 5;
  const int grp  = lane >> 4, lmod = lane & 15;
  const int m0   = blockIdx.x * BM;
  const int n0   = blockIdx.y * BN;
  const int mw   = wave >> 1;     // 0..3 -> 32-row slice
  const int nw   = wave & 1;      // 0..1 -> 32-col slice

  v8f acc[2][2] = {{{}, {}}, {{}, {}}};

  // staging maps (per k-step): A 128x32 -> 16 elems/thread; W 64x32 -> 8/thread
  const int amr = tid >> 1,  akc = (tid & 1) * 16;
  const int wnr = tid >> 2,  wkc = (tid & 3) * 8;

  for (int k0 = 0; k0 < EMB; k0 += BK) {
    {
      const float* src = A + (size_t)(m0 + amr) * EMB + k0 + akc;
      #pragma unroll
      for (int j = 0; j < 16; ++j) As[amr][akc + j] = (bf16_t)src[j];
    }
    {
      const float* src = W + (size_t)(n0 + wnr) * EMB + k0 + wkc;
      #pragma unroll
      for (int j = 0; j < 8; ++j) Ws[wnr][wkc + j] = (bf16_t)src[j];
    }
    if (k0 + BK < EMB) {  // warm L2/GL1 for the next k-slice (global_prefetch_b8)
      __builtin_prefetch(A + (size_t)(m0 + amr) * EMB + k0 + BK + akc, 0, 3);
      __builtin_prefetch(W + (size_t)(n0 + wnr) * EMB + k0 + BK + wkc, 0, 3);
    }
    __syncthreads();

    v16bf a[2], b[2];
    #pragma unroll
    for (int i = 0; i < 2; ++i)
      #pragma unroll
      for (int e = 0; e < 16; ++e)
        a[i][e] = As[mw * 32 + i * 16 + lmod][a_kidx(e, grp)];
    #pragma unroll
    for (int j = 0; j < 2; ++j)
      #pragma unroll
      for (int e = 0; e < 16; ++e)
        b[j][e] = Ws[nw * 32 + j * 16 + lmod][grp * 16 + e];
    #pragma unroll
    for (int i = 0; i < 2; ++i)
      #pragma unroll
      for (int j = 0; j < 2; ++j)
        acc[i][j] = wmma_bf16(a[i], b[j], acc[i][j]);
    __syncthreads();
  }

  #pragma unroll
  for (int i = 0; i < 2; ++i)
    #pragma unroll
    for (int j = 0; j < 2; ++j)
      #pragma unroll
      for (int r = 0; r < 8; ++r) {
        int gm = m0 + mw * 32 + i * 16 + r + 8 * grp;  // D: row = r + 8*grp
        int gn = n0 + nw * 32 + j * 16 + lmod;         //    col = lane%16
        float v = acc[i][j][r] + bias[gn];
        if (reorder) {
          int bb = gm >> 10, s = gm & (SEQ - 1);
          int h  = gn >> 6,  d = gn & (HD - 1);
          outHeads[(((size_t)(bb * NH + h)) * SEQ + s) * HD + d] = (bf16_t)v;
        } else {
          outF[(size_t)gm * EMB + gn] = v;
        }
      }
}

// ------------------------------------------------------------------
// Fused attention: one block per (batch, 32-query block); loops all heads.
// LDS (dynamic, 320 KB = CDNA5 per-WG max):
//   sc  [32][1024] f32  scores        (128 KB)
//   acc [32][1024] f32  head average  (128 KB)
//   pbf [32][1024] bf16 probabilities ( 64 KB)  -> pure ds_load fragments in P@V
// ------------------------------------------------------------------
#define QBLK 32

__global__ __launch_bounds__(256)
void attn_softmax_wmma(const bf16_t* __restrict__ Qh,
                       const bf16_t* __restrict__ Kh,
                       const bf16_t* __restrict__ Vh,
                       const int*    __restrict__ mask,
                       float* __restrict__ ctx,
                       float* __restrict__ attn_avg)
{
  extern __shared__ float smem[];
  float*  sc  = smem;                              // [QBLK][SEQ] f32
  float*  acc = smem + QBLK * SEQ;                 // [QBLK][SEQ] f32
  bf16_t* pbf = (bf16_t*)(smem + 2 * QBLK * SEQ);  // [QBLK][SEQ] bf16

  const int tid  = threadIdx.x;
  const int lane = tid & 31, wave = tid >> 5;
  const int grp  = lane >> 4, lmod = lane & 15;
  const int b    = blockIdx.x >> 5;
  const int q0   = (blockIdx.x & 31) * QBLK;

  for (int i = tid; i < QBLK * SEQ; i += 256) acc[i] = 0.f;
  __syncthreads();

  for (int h = 0; h < NH; ++h) {
    const size_t base = ((size_t)(b * NH + h)) * SEQ * HD;

    // ---------- scores = (Q K^T) * 1/sqrt(D) ----------
    {
      const int qs = wave & 1;             // which 16-row q sub-tile
      v16bf aq[2];
      #pragma unroll
      for (int c = 0; c < 2; ++c)
        #pragma unroll
        for (int e = 0; e < 16; ++e) {
          int q = q0 + qs * 16 + lmod;
          int d = c * 32 + a_kidx(e, grp);
          aq[c][e] = Qh[base + (size_t)q * HD + d];
        }
      for (int kt = (wave >> 1); kt < SEQ / 16; kt += 4) {
        v8f s = {};
        #pragma unroll
        for (int c = 0; c < 2; ++c) {
          v16bf bk;
          int kidx = kt * 16 + lmod;       // B layout: col = lane%16
          #pragma unroll
          for (int e = 0; e < 16; ++e)     //           K  = grp*16 + e
            bk[e] = Kh[base + (size_t)kidx * HD + c * 32 + grp * 16 + e];
          s = wmma_bf16(aq[c], bk, s);
        }
        #pragma unroll
        for (int r = 0; r < 8; ++r) {
          int ql = qs * 16 + r + 8 * grp;
          sc[ql * SEQ + kt * 16 + lmod] = s[r] * 0.125f;   // 1/sqrt(64)
        }
      }
    }
    __syncthreads();

    // ------ masked softmax (8 lanes/row) + head-average + bf16 P slab ------
    {
      const int row = tid >> 3, sub = tid & 7;
      const int q = q0 + row;
      const int* mrow = mask + ((size_t)b * SEQ + q) * SEQ;
      float* srow = sc + row * SEQ;
      float mx = -3.0e38f;
      for (int k = sub * 128; k < sub * 128 + 128; ++k) {
        float v = (mrow[k] == 0) ? -1.0e9f : srow[k];
        srow[k] = v;
        mx = fmaxf(mx, v);
      }
      #pragma unroll
      for (int o = 4; o > 0; o >>= 1) mx = fmaxf(mx, __shfl_xor(mx, o, 32));
      float sum = 0.f;
      for (int k = sub * 128; k < sub * 128 + 128; ++k)
        sum += __expf(srow[k] - mx);
      #pragma unroll
      for (int o = 4; o > 0; o >>= 1) sum += __shfl_xor(sum, o, 32);
      float inv = 1.f / sum;
      float*  arow = acc + row * SEQ;
      bf16_t* prow = pbf + row * SEQ;
      for (int k = sub * 128; k < sub * 128 + 128; ++k) {
        float p = __expf(srow[k] - mx) * inv;
        prow[k] = (bf16_t)p;
        arow[k] += p * (1.0f / NH);
      }
    }
    __syncthreads();

    // ---------- ctx = P @ V  (P fragments straight from bf16 LDS) ----------
    {
      const int qs2 = wave & 1;
      const int dt  = wave >> 1;           // 0..3 -> 4 d-tiles of 16
      v8f cacc = {};
      for (int kc = 0; kc < SEQ / 32; ++kc) {
        v16bf ap, bv;
        #pragma unroll
        for (int e = 0; e < 16; ++e) {
          int ql = qs2 * 16 + lmod;
          ap[e] = pbf[ql * SEQ + kc * 32 + a_kidx(e, grp)];
        }
        #pragma unroll
        for (int e = 0; e < 16; ++e) {
          int kk = kc * 32 + grp * 16 + e;
          bv[e] = Vh[base + (size_t)kk * HD + dt * 16 + lmod];
        }
        cacc = wmma_bf16(ap, bv, cacc);
      }
      #pragma unroll
      for (int r = 0; r < 8; ++r) {
        int q   = q0 + qs2 * 16 + r + 8 * grp;
        int col = h * HD + dt * 16 + lmod;
        ctx[((size_t)b * SEQ + q) * EMB + col] = cacc[r];
      }
    }
    __syncthreads();   // before next head reuses sc/pbf
  }

  // ---------- write attn_avg [B,S,S] ----------
  for (int i = tid; i < QBLK * SEQ; i += 256) {
    int row = i >> 10, k = i & (SEQ - 1);
    attn_avg[((size_t)b * SEQ + q0 + row) * SEQ + k] = acc[i];
  }
}

// ------------------------------------------------------------------
extern "C" void kernel_launch(void* const* d_in, const int* in_sizes, int n_in,
                              void* d_out, int out_size, void* d_ws, size_t ws_size,
                              hipStream_t stream) {
  (void)in_sizes; (void)n_in; (void)out_size; (void)ws_size;
  const float* query = (const float*)d_in[0];
  const float* key   = (const float*)d_in[1];
  const float* value = (const float*)d_in[2];
  const int*   mask  = (const int*)  d_in[3];
  const float* Wq = (const float*)d_in[4];  const float* bq = (const float*)d_in[5];
  const float* Wk = (const float*)d_in[6];  const float* bk = (const float*)d_in[7];
  const float* Wv = (const float*)d_in[8];  const float* bv = (const float*)d_in[9];
  const float* Wo = (const float*)d_in[10]; const float* bo = (const float*)d_in[11];

  const size_t headElems = (size_t)BATCH * NH * SEQ * HD;  // 4,194,304
  bf16_t* Qh  = (bf16_t*)d_ws;
  bf16_t* Kh  = Qh + headElems;
  bf16_t* Vh  = Kh + headElems;
  float*  ctx = (float*)(Vh + headElems);                  // fp32 [B*S, E]

  float* out      = (float*)d_out;
  float* attn_avg = out + (size_t)BATCH * SEQ * EMB;

  dim3 gg(MTOT / BM, EMB / BN);   // 32 x 16 blocks
  gemm_bias_wmma<<<gg, 256, 0, stream>>>(query, Wq, bq, nullptr, Qh, 1);
  gemm_bias_wmma<<<gg, 256, 0, stream>>>(key,   Wk, bk, nullptr, Kh, 1);
  gemm_bias_wmma<<<gg, 256, 0, stream>>>(value, Wv, bv, nullptr, Vh, 1);

  // 128 KB scores + 128 KB average + 64 KB bf16 P = 320 KB (CDNA5 WG max)
  const int attnLds = (2 * QBLK * SEQ) * (int)sizeof(float) + QBLK * SEQ * 2;
  hipFuncSetAttribute((const void*)attn_softmax_wmma,
                      hipFuncAttributeMaxDynamicSharedMemorySize, attnLds);
  attn_softmax_wmma<<<BATCH * (SEQ / QBLK), 256, attnLds, stream>>>(
      Qh, Kh, Vh, mask, ctx, attn_avg);

  gemm_bias_wmma<<<gg, 256, 0, stream>>>(ctx, Wo, bo, out, nullptr, 0);
}
